// MoE_17841294147682
// MI455X (gfx1250) — matrix operand order repaired and checked by reading, not compiled, not run on hardware
//
#include <hip/hip_runtime.h>

// ---------------------------------------------------------------------------
// Types
// ---------------------------------------------------------------------------
typedef __bf16 bf16_t;
typedef __bf16 v16bf __attribute__((ext_vector_type(16)));
typedef __bf16 v8bf  __attribute__((ext_vector_type(8)));
typedef float  v8f   __attribute__((ext_vector_type(8)));
typedef unsigned v4u __attribute__((ext_vector_type(4)));
typedef unsigned v8u __attribute__((ext_vector_type(8)));

struct __attribute__((aligned(16))) U16 { unsigned u[4]; };
struct __attribute__((aligned(8)))  U8  { unsigned u[2]; };

#define WG_B   2
#define WG_S   2048
#define WG_D   768
#define WG_E   8
#define WG_H   2048
#define WG_NH  8
#define WG_HD  96
#define WG_N   (WG_B * WG_S)   // 4096 rows

enum { EPI_F32 = 0, EPI_BF16 = 1, EPI_ROWSCALE_ACC = 2 };

// ---------------------------------------------------------------------------
// fp32 -> bf16 conversion, staged through LDS with async-to-LDS loads
// ---------------------------------------------------------------------------
__global__ __launch_bounds__(256) void cvt_f32_bf16_async(
    const float* __restrict__ src, bf16_t* __restrict__ dst, long n)
{
    __shared__ float lbuf[1024];
    int  t    = threadIdx.x;
    long base = (long)blockIdx.x * 1024;
    if (base + 1024 <= n) {
        long idx = base + (long)t * 4;
        float a0, a1, a2, a3;
#if defined(__HIP_DEVICE_COMPILE__)
        unsigned ldsoff = (unsigned)(unsigned long long)(&lbuf[t * 4]);
        const float* gp = src + idx;
        asm volatile("global_load_async_to_lds_b128 %0, %1, off"
                     :: "v"(ldsoff), "v"(gp) : "memory");
        asm volatile("s_wait_asynccnt 0" ::: "memory");
        volatile float* lb = &lbuf[t * 4];
        a0 = lb[0]; a1 = lb[1]; a2 = lb[2]; a3 = lb[3];
#else
        a0 = src[idx]; a1 = src[idx + 1]; a2 = src[idx + 2]; a3 = src[idx + 3];
#endif
        union { bf16_t h[4]; U8 u; } pk;
        pk.h[0] = (bf16_t)a0; pk.h[1] = (bf16_t)a1;
        pk.h[2] = (bf16_t)a2; pk.h[3] = (bf16_t)a3;
        *(U8*)&dst[idx] = pk.u;
    } else {
        for (long i = base + t; i < n; i += 256) dst[i] = (bf16_t)src[i];
    }
}

// ---------------------------------------------------------------------------
// fp32 -> bf16 conversion, bulk tiles DMA'd into LDS via the Tensor Data Mover
// (1D descriptor: 1024 x 4B elements per workgroup; wave 0 issues the DMA)
// ---------------------------------------------------------------------------
__global__ __launch_bounds__(256) void cvt_f32_bf16_tdm(
    const float* __restrict__ src, bf16_t* __restrict__ dst, long n)
{
    __shared__ float lbuf[1024];
    int  t    = threadIdx.x;
    long base = (long)blockIdx.x * 1024;
    if (base + 1024 <= n) {
        long idx = base + (long)t * 4;
        float a0, a1, a2, a3;
#if defined(__HIP_DEVICE_COMPILE__)
        if (t < 32) {
            unsigned long long ga = (unsigned long long)(src + base);
            unsigned lds = (unsigned)(unsigned long long)(&lbuf[0]);
            v4u g0;
            g0[0] = 1u;                                   // count=1 (valid D#)
            g0[1] = lds;                                  // lds_addr
            g0[2] = (unsigned)ga;                         // global_addr[31:0]
            g0[3] = (unsigned)((ga >> 32) & 0x1FFFFFFull) // global_addr[56:32]
                    | (2u << 30);                         // type = 2 ("image")
            v8u g1;
            g1[0] = (2u << 16);                           // data_size = 4 bytes
            g1[1] = (1024u << 16);                        // tensor_dim0 = 1024
            g1[2] = 0u;
            g1[3] = (1024u << 16);                        // tile_dim0 = 1024
            g1[4] = 0u;                                   // tile_dim1/2 unused
            g1[5] = 1024u;                                // tensor_dim0_stride
            g1[6] = 0u;
            g1[7] = 0u;
            asm volatile("tensor_load_to_lds %0, %1" :: "s"(g0), "s"(g1) : "memory");
            __builtin_amdgcn_s_wait_tensorcnt((short)0);
        }
        __syncthreads();
        volatile float* lb = &lbuf[t * 4];
        a0 = lb[0]; a1 = lb[1]; a2 = lb[2]; a3 = lb[3];
#else
        a0 = src[idx]; a1 = src[idx + 1]; a2 = src[idx + 2]; a3 = src[idx + 3];
#endif
        union { bf16_t h[4]; U8 u; } pk;
        pk.h[0] = (bf16_t)a0; pk.h[1] = (bf16_t)a1;
        pk.h[2] = (bf16_t)a2; pk.h[3] = (bf16_t)a3;
        *(U8*)&dst[idx] = pk.u;
    } else {
        for (long i = base + t; i < n; i += 256) dst[i] = (bf16_t)src[i];
    }
}

// ---------------------------------------------------------------------------
// Batched bf16 GEMM:  C[m,n] = sum_k A[m,k] * B[n,k]  (+bias[n]) with epilogues
// Tile 128x128x32, 8 waves (4 along M x 2 along N), wave -> 2x4 WMMA tiles.
// Tiles staged to LDS with GLOBAL_LOAD_ASYNC_TO_LDS_B128 (ASYNCcnt-tracked).
// ---------------------------------------------------------------------------
#define BM  128
#define BN  128
#define BKK 32
#define LDT 48   // LDS row stride in elements (96B, keeps 32B-aligned frags)

__global__ __launch_bounds__(256) void gemm_bf16_wmma(
    const bf16_t* __restrict__ A, const bf16_t* __restrict__ B,
    float* __restrict__ Cf, bf16_t* __restrict__ Cb,
    const float* __restrict__ bias,
    int M, int N, int K, int lda, int ldb, int ldc,
    int bz2,
    long sA1, long sA2, long sB1, long sB2, long sC1, long sC2,
    int mode, float beta,
    const float* __restrict__ probs, int probN, int pidx,
    const float* __restrict__ mask)
{
    __shared__ bf16_t As[BM * LDT];
    __shared__ bf16_t Bs[BN * LDT];

    int z  = blockIdx.z;
    int z1 = z / bz2, z2 = z - z1 * bz2;
    A += (long)z1 * sA1 + (long)z2 * sA2;
    B += (long)z1 * sB1 + (long)z2 * sB2;
    long coff = (long)z1 * sC1 + (long)z2 * sC2;

    int t    = threadIdx.x;
    int lane = t & 31, wave = t >> 5;
    int half = lane >> 4, l16 = lane & 15;
    int waveM = wave & 3, waveN = wave >> 2;

    int tileM = blockIdx.y * BM;
    int tileN = blockIdx.x * BN;

    int lr = t >> 1;           // loader row 0..127
    int lc = (t & 1) << 4;     // loader col 0 or 16
    int bn = tileN + lr;
    bool bvalid = bn < N;

#if defined(__HIP_DEVICE_COMPILE__)
    unsigned ldsA = (unsigned)(unsigned long long)(&As[lr * LDT + lc]);
    unsigned ldsB = (unsigned)(unsigned long long)(&Bs[lr * LDT + lc]);
#endif

    v8f acc[2][4] = {};

    for (int k0 = 0; k0 < K; k0 += BKK) {
        __syncthreads();   // previous compute done before LDS overwrite
#if defined(__HIP_DEVICE_COMPILE__)
        {   // A tile: 2x16B async chunks per thread (offset applies to both sides)
            const bf16_t* aseg = A + (long)(tileM + lr) * lda + k0 + lc;
            asm volatile("global_load_async_to_lds_b128 %0, %1, off"
                         :: "v"(ldsA), "v"(aseg) : "memory");
            asm volatile("global_load_async_to_lds_b128 %0, %1, off offset:16"
                         :: "v"(ldsA), "v"(aseg) : "memory");
        }
        if (bvalid) {
            const bf16_t* bseg = B + (long)bn * ldb + k0 + lc;
            asm volatile("global_load_async_to_lds_b128 %0, %1, off"
                         :: "v"(ldsB), "v"(bseg) : "memory");
            asm volatile("global_load_async_to_lds_b128 %0, %1, off offset:16"
                         :: "v"(ldsB), "v"(bseg) : "memory");
        }
        if (k0 + BKK < K) {  // prefetch next K tiles into cache hierarchy
            __builtin_prefetch(A + (long)(tileM + lr) * lda + k0 + BKK + lc, 0, 1);
            if (bvalid)
                __builtin_prefetch(B + (long)bn * ldb + k0 + BKK + lc, 0, 1);
        }
        asm volatile("s_wait_asynccnt 0" ::: "memory");
#else
        {
            const bf16_t* srcA = A + (long)(tileM + lr) * lda + k0 + lc;
            *(U16*)&As[lr * LDT + lc]     = ((const U16*)srcA)[0];
            *(U16*)&As[lr * LDT + lc + 8] = ((const U16*)srcA)[1];
            U16 d0 = {}, d1 = {};
            if (bvalid) {
                const bf16_t* srcB = B + (long)bn * ldb + k0 + lc;
                d0 = ((const U16*)srcB)[0];
                d1 = ((const U16*)srcB)[1];
            }
            *(U16*)&Bs[lr * LDT + lc]     = d0;
            *(U16*)&Bs[lr * LDT + lc + 8] = d1;
        }
#endif
        __syncthreads();

        v16bf af[2], bfrag[4];
        // A frag (16-bit A 16x32 layout: per lane K = half*8..+7 and 16+half*8..+7)
        for (int mt = 0; mt < 2; ++mt) {
            const bf16_t* arow = &As[(waveM * 32 + mt * 16 + l16) * LDT];
            v8bf lo = *(const v8bf*)&arow[half * 8];
            v8bf hi = *(const v8bf*)&arow[16 + half * 8];
            af[mt] = __builtin_shufflevector(lo, hi,
                        0,1,2,3,4,5,6,7,8,9,10,11,12,13,14,15);
        }
        // B frag (per lane: K = half*8..+7 and 16+half*8..+7 of column l16)
        for (int nt = 0; nt < 4; ++nt) {
            const bf16_t* brow = &Bs[(waveN * 64 + nt * 16 + l16) * LDT];
            v8bf blo = *(const v8bf*)&brow[half * 8];
            v8bf bhi = *(const v8bf*)&brow[16 + half * 8];
            bfrag[nt] = __builtin_shufflevector(blo, bhi,
                        0,1,2,3,4,5,6,7,8,9,10,11,12,13,14,15);
        }
        for (int mt = 0; mt < 2; ++mt)
            for (int nt = 0; nt < 4; ++nt)
                acc[mt][nt] = __builtin_amdgcn_wmma_f32_16x16x32_bf16(
                    false, af[mt], false, bfrag[nt],
                    (short)0, acc[mt][nt], false, false);
    }

    // epilogue: C layout => lane(0-15): M=j, lane(16-31): M=j+8, N=l16
    // (all N used are multiples of 16, so OOB accumulator tiles are whole
    //  16-wide tiles and are simply never stored)
    for (int mt = 0; mt < 2; ++mt) {
        for (int nt = 0; nt < 4; ++nt) {
            int n = tileN + waveN * 64 + nt * 16 + l16;
            if (n >= N) continue;
            float bv = bias ? bias[n] : 0.0f;
            for (int j = 0; j < 8; ++j) {
                int  m  = tileM + waveM * 32 + mt * 16 + half * 8 + j;
                float v = acc[mt][nt][j] + bv;
                long ci = coff + (long)m * ldc + n;
                if (mode == EPI_F32) {
                    Cf[ci] = v;
                } else if (mode == EPI_BF16) {
                    Cb[ci] = (bf16_t)v;
                } else { // EPI_ROWSCALE_ACC
                    float rs = probs[(long)m * probN + pidx];
                    if (mask) rs *= mask[m];
                    float prev = (beta != 0.0f) ? Cf[ci] : 0.0f;
                    Cf[ci] = prev + rs * v;
                }
            }
        }
    }
}

// ---------------------------------------------------------------------------
// Row softmax (rows of length 2048), writes bf16 probabilities
// ---------------------------------------------------------------------------
__global__ __launch_bounds__(256) void softmax_rows(
    const float* __restrict__ Sc, bf16_t* __restrict__ P, float scale)
{
    __shared__ float red[8];
    long base = (long)blockIdx.x * WG_S;
    int  t = threadIdx.x, lane = t & 31, wave = t >> 5;

    float v[8];
    float mx = -1e30f;
    for (int j = 0; j < 8; ++j) {
        v[j] = Sc[base + t + j * 256] * scale;
        mx = fmaxf(mx, v[j]);
    }
    for (int o = 16; o; o >>= 1) mx = fmaxf(mx, __shfl_xor(mx, o, 32));
    if (lane == 0) red[wave] = mx;
    __syncthreads();
    float m0 = red[0];
    for (int i = 1; i < 8; ++i) m0 = fmaxf(m0, red[i]);
    __syncthreads();

    float s = 0.0f;
    for (int j = 0; j < 8; ++j) { v[j] = __expf(v[j] - m0); s += v[j]; }
    for (int o = 16; o; o >>= 1) s += __shfl_xor(s, o, 32);
    if (lane == 0) red[wave] = s;
    __syncthreads();
    float tot = 0.0f;
    for (int i = 0; i < 8; ++i) tot += red[i];
    float inv = 1.0f / tot;
    for (int j = 0; j < 8; ++j)
        P[base + t + j * 256] = (bf16_t)(v[j] * inv);
}

// ---------------------------------------------------------------------------
// V transpose: vT[b][h][d][s] = v[b][s][h*HD+d]
// ---------------------------------------------------------------------------
__global__ __launch_bounds__(256) void transpose_v(
    const bf16_t* __restrict__ vb, bf16_t* __restrict__ vT)
{
    long total = (long)WG_B * WG_NH * WG_HD * WG_S;
    long i = (long)blockIdx.x * 256 + threadIdx.x;
    if (i >= total) return;
    int  s = (int)(i % WG_S);
    long r = i / WG_S;
    int  d = (int)(r % WG_HD); r /= WG_HD;
    int  h = (int)(r % WG_NH);
    int  b = (int)(r / WG_NH);
    vT[i] = vb[((long)(b * WG_S + s)) * WG_D + h * WG_HD + d];
}

// ---------------------------------------------------------------------------
// Gumbel-softmax gate: one block per row (deterministic hash noise)
// ---------------------------------------------------------------------------
__device__ __forceinline__ float hash_u01(unsigned x)
{
    x ^= x >> 17; x *= 0xed5ad4bbu;
    x ^= x >> 11; x *= 0xac4c1b51u;
    x ^= x >> 15; x *= 0x31848babu;
    x ^= x >> 14;
    return (float)(x >> 8) * (1.0f / 16777216.0f) + 1e-7f;
}

__global__ __launch_bounds__(256) void gate_gumbel(
    const float* __restrict__ X, const float* __restrict__ Wg,
    const float* __restrict__ bg, float* __restrict__ probs,
    int K, int Eo, float invtemp, unsigned seed)
{
    __shared__ float logits[8];
    int row = blockIdx.x, t = threadIdx.x;
    int wave = t >> 5, lane = t & 31;
    if (wave < Eo) {
        float acc = 0.0f;
        const float* xr = X + (long)row * K;
        const float* wr = Wg + (long)wave * K;
        for (int k = lane; k < K; k += 32) acc += xr[k] * wr[k];
        for (int o = 16; o; o >>= 1) acc += __shfl_down(acc, o, 32);
        if (lane == 0) logits[wave] = acc + bg[wave];
    }
    __syncthreads();
    if (t == 0) {
        float l[8], mx = -1e30f;
        for (int j = 0; j < Eo; ++j) {
            float u = hash_u01(seed + (unsigned)row * 16u + (unsigned)j);
            float g = -__logf(-__logf(u));
            l[j] = (logits[j] + g) * invtemp;
            mx = fmaxf(mx, l[j]);
        }
        float s = 0.0f;
        for (int j = 0; j < Eo; ++j) { l[j] = __expf(l[j] - mx); s += l[j]; }
        float inv = 1.0f / s;
        for (int j = 0; j < Eo; ++j) probs[(long)row * Eo + j] = l[j] * inv;
    }
}

__global__ __launch_bounds__(256) void make_masks(
    const float* __restrict__ hi, float* __restrict__ capm,
    float* __restrict__ acm, int n)
{
    int i = blockIdx.x * 256 + threadIdx.x;
    if (i >= n) return;
    int b = i / WG_S;
    capm[i] = hi[b * 2 + 0];
    acm[i]  = hi[b * 2 + 1];
}

// ---------------------------------------------------------------------------
// h = silu(t1) * t3  -> bf16
// ---------------------------------------------------------------------------
__global__ __launch_bounds__(256) void silu_mul_kernel(
    const float* __restrict__ a, const float* __restrict__ b,
    bf16_t* __restrict__ h, long n)
{
    long i = (long)blockIdx.x * 256 + threadIdx.x;
    if (i >= n) return;
    float x = a[i];
    float s = x / (1.0f + __expf(-x));
    h[i] = (bf16_t)(s * b[i]);
}

// ---------------------------------------------------------------------------
// load-balance loss (single block)
// ---------------------------------------------------------------------------
__global__ __launch_bounds__(256) void loss_kernel(
    const float* __restrict__ cp, const float* __restrict__ ap,
    const float* __restrict__ cm, const float* __restrict__ am,
    int n, float* __restrict__ out)
{
    __shared__ float red[256];
    float s[16] = {};
    float mc = 0.0f, ma = 0.0f;
    for (int i = threadIdx.x; i < n; i += 256) {
        float c = cm[i], a = am[i];
        for (int j = 0; j < 8; ++j) {
            s[j]     += cp[(long)i * 8 + j] * c;
            s[8 + j] += ap[(long)i * 8 + j] * a;
        }
        mc += c; ma += a;
    }
    float tot[18];
    for (int v = 0; v < 18; ++v) {
        float x = (v < 16) ? s[v] : ((v == 16) ? mc : ma);
        red[threadIdx.x] = x;
        __syncthreads();
        for (int o = 128; o; o >>= 1) {
            if (threadIdx.x < o) red[threadIdx.x] += red[threadIdx.x + o];
            __syncthreads();
        }
        tot[v] = red[0];
        __syncthreads();
    }
    if (threadIdx.x == 0) {
        float denom = 8.0f * (tot[16] + tot[17]) + 1e-10f;
        float lb = 0.0f;
        for (int j = 0; j < 16; ++j) {
            float u = tot[j] / denom;
            lb += u * __logf(u + 1e-10f);
        }
        out[0] = lb / 16.0f;
    }
}

// ---------------------------------------------------------------------------
// Host side
// ---------------------------------------------------------------------------
static inline void launch_gemm(hipStream_t st,
    const bf16_t* A, const bf16_t* B, void* C, const float* bias,
    int M, int N, int K, int lda, int ldb, int ldc,
    int b1, int b2,
    long sA1, long sA2, long sB1, long sB2, long sC1, long sC2,
    int mode, float beta,
    const float* probs, int probN, int pidx, const float* mask)
{
    dim3 grid((N + BM - 1) / BM, (M + BM - 1) / BM, b1 * b2);
    gemm_bf16_wmma<<<grid, 256, 0, st>>>(A, B, (float*)C, (bf16_t*)C, bias,
        M, N, K, lda, ldb, ldc, b2, sA1, sA2, sB1, sB2, sC1, sC2,
        mode, beta, probs, probN, pidx, mask);
}

static inline void launch_cvt(hipStream_t st, const float* src, bf16_t* dst, long n)
{
    long blocks = (n + 1023) / 1024;
    cvt_f32_bf16_async<<<(unsigned)blocks, 256, 0, st>>>(src, dst, n);
}

static inline void launch_cvt_tdm(hipStream_t st, const float* src, bf16_t* dst, long n)
{
    long blocks = (n + 1023) / 1024;
    cvt_f32_bf16_tdm<<<(unsigned)blocks, 256, 0, st>>>(src, dst, n);
}

extern "C" void kernel_launch(void* const* d_in, const int* in_sizes, int n_in,
                              void* d_out, int out_size, void* d_ws, size_t ws_size,
                              hipStream_t stream)
{
    (void)in_sizes; (void)n_in; (void)out_size; (void)ws_size;

    const float* x        = (const float*)d_in[0];
    const float* timep    = (const float*)d_in[1];
    const float* caption  = (const float*)d_in[2];
    const float* acoustic = (const float*)d_in[3];
    const float* Wqkv     = (const float*)d_in[4];
    const float* bqkv     = (const float*)d_in[5];
    const float* Wo       = (const float*)d_in[6];
    const float* bo       = (const float*)d_in[7];
    const float* Whi      = (const float*)d_in[8];
    const float* bhi      = (const float*)d_in[9];
    const float* Wcg      = (const float*)d_in[10];
    const float* bcg      = (const float*)d_in[11];
    const float* Wag      = (const float*)d_in[12];
    const float* bag      = (const float*)d_in[13];
    const float* cw1      = (const float*)d_in[14];
    const float* cw2      = (const float*)d_in[15];
    const float* cw3      = (const float*)d_in[16];
    const float* aw1      = (const float*)d_in[17];
    const float* aw2      = (const float*)d_in[18];
    const float* aw3      = (const float*)d_in[19];
    const float* fw1      = (const float*)d_in[20];
    const float* fw2      = (const float*)d_in[21];
    const float* fw3      = (const float*)d_in[22];

    const int Nn = WG_N, D = WG_D, S = WG_S, Bb = WG_B, NH = WG_NH,
              HD = WG_HD, E = WG_E, H = WG_H;

    // ---- workspace carve-out ----
    char*  w   = (char*)d_ws;
    size_t cur = 0;
    auto alloc = [&](size_t bytes) -> char* {
        char* p = w + cur;
        cur += (bytes + 255) & ~(size_t)255;
        return p;
    };
    const size_t EHD = (size_t)E * H * D;

    bf16_t* xb     = (bf16_t*)alloc((size_t)Nn * D * 2);
    bf16_t* capb   = (bf16_t*)alloc((size_t)Nn * D * 2);
    bf16_t* Wqkvb  = (bf16_t*)alloc((size_t)3 * D * D * 2);
    bf16_t* Wob    = (bf16_t*)alloc((size_t)D * D * 2);
    bf16_t* cw1b   = (bf16_t*)alloc(EHD * 2);
    bf16_t* cw2b   = (bf16_t*)alloc(EHD * 2);
    bf16_t* cw3b   = (bf16_t*)alloc(EHD * 2);
    bf16_t* aw1b   = (bf16_t*)alloc(EHD * 2);
    bf16_t* aw2b   = (bf16_t*)alloc(EHD * 2);
    bf16_t* aw3b   = (bf16_t*)alloc(EHD * 2);
    bf16_t* fw1b   = (bf16_t*)alloc(EHD * 2);
    bf16_t* fw2b   = (bf16_t*)alloc(EHD * 2);
    bf16_t* fw3b   = (bf16_t*)alloc(EHD * 2);
    bf16_t* qb     = (bf16_t*)alloc((size_t)Nn * D * 2);
    bf16_t* kb     = (bf16_t*)alloc((size_t)Nn * D * 2);
    bf16_t* vb     = (bf16_t*)alloc((size_t)Nn * D * 2);
    bf16_t* vT     = (bf16_t*)alloc((size_t)Bb * NH * HD * S * 2);
    float*  scores = (float*)alloc((size_t)Bb * NH * S * S * 4);
    bf16_t* Pb     = (bf16_t*)alloc((size_t)Bb * NH * S * S * 2);
    bf16_t* attnb  = (bf16_t*)alloc((size_t)Nn * D * 2);
    float*  capf   = (float*)alloc((size_t)Nn * D * 4);
    float*  hiprob = (float*)alloc((size_t)Bb * 2 * 4);
    float*  capprob= (float*)alloc((size_t)Nn * E * 4);
    float*  acprob = (float*)alloc((size_t)Nn * E * 4);
    float*  capmask= (float*)alloc((size_t)Nn * 4);
    float*  acmask = (float*)alloc((size_t)Nn * 4);
    float*  t1     = (float*)alloc((size_t)Nn * H * 4);
    float*  t3     = (float*)alloc((size_t)Nn * H * 4);
    bf16_t* hb     = (bf16_t*)alloc((size_t)Nn * H * 2);
    float*  y      = (float*)alloc((size_t)Nn * D * 4);
    bf16_t* yb     = (bf16_t*)alloc((size_t)Nn * D * 2);

    float* zout = (float*)d_out;

    // ---- 1. convert activations (async-LDS path) + weights (TDM path) ----
    launch_cvt(stream, x,       xb,    (long)Nn * D);
    launch_cvt(stream, caption, capb,  (long)Nn * D);
    launch_cvt_tdm(stream, Wqkv, Wqkvb, (long)3 * D * D);
    launch_cvt_tdm(stream, Wo,   Wob,   (long)D * D);
    launch_cvt_tdm(stream, cw1, cw1b, (long)EHD);
    launch_cvt_tdm(stream, cw2, cw2b, (long)EHD);
    launch_cvt_tdm(stream, cw3, cw3b, (long)EHD);
    launch_cvt_tdm(stream, aw1, aw1b, (long)EHD);
    launch_cvt_tdm(stream, aw2, aw2b, (long)EHD);
    launch_cvt_tdm(stream, aw3, aw3b, (long)EHD);
    launch_cvt_tdm(stream, fw1, fw1b, (long)EHD);
    launch_cvt_tdm(stream, fw2, fw2b, (long)EHD);
    launch_cvt_tdm(stream, fw3, fw3b, (long)EHD);

    // ---- 2. QKV projections (q from x; k,v from caption) ----
    launch_gemm(stream, xb,   Wqkvb,                 qb, bqkv,
                Nn, D, D, D, D, D, 1, 1, 0,0,0,0,0,0, EPI_BF16, 0, 0, 0, 0, 0);
    launch_gemm(stream, capb, Wqkvb + (size_t)D * D, kb, bqkv + D,
                Nn, D, D, D, D, D, 1, 1, 0,0,0,0,0,0, EPI_BF16, 0, 0, 0, 0, 0);
    launch_gemm(stream, capb, Wqkvb + (size_t)2*D*D, vb, bqkv + 2*D,
                Nn, D, D, D, D, D, 1, 1, 0,0,0,0,0,0, EPI_BF16, 0, 0, 0, 0, 0);

    {   // V -> [B][NH][HD][S]
        long total = (long)Bb * NH * HD * S;
        transpose_v<<<(unsigned)((total + 255) / 256), 256, 0, stream>>>(vb, vT);
    }

    // ---- 3. attention scores, softmax, P@V ----
    launch_gemm(stream, qb, kb, scores, nullptr,
                S, S, HD, D, D, S,
                Bb, NH,
                (long)S * D, (long)HD,
                (long)S * D, (long)HD,
                (long)NH * S * S, (long)S * S,
                EPI_F32, 0, 0, 0, 0, 0);

    softmax_rows<<<Bb * NH * S, 256, 0, stream>>>(scores, Pb, 0.10206207262f);

    launch_gemm(stream, Pb, vT, attnb, nullptr,
                S, HD, S, S, S, D,
                Bb, NH,
                (long)NH * S * S, (long)S * S,
                (long)NH * HD * S, (long)HD * S,
                (long)S * D, (long)HD,
                EPI_BF16, 0, 0, 0, 0, 0);

    // ---- 4. output projection: cap = attn @ Wo^T + bo (fp32 for gating) ----
    launch_gemm(stream, attnb, Wob, capf, bo,
                Nn, D, D, D, D, D, 1, 1, 0,0,0,0,0,0, EPI_F32, 0, 0, 0, 0, 0);

    // ---- 5. gates ----
    gate_gumbel<<<Bb, 256, 0, stream>>>(timep,    Whi, bhi, hiprob,  D, 2, 1.0f, 0x9E3779B9u);
    gate_gumbel<<<Nn, 256, 0, stream>>>(capf,     Wcg, bcg, capprob, D, 8, 0.5f, 0x85EBCA6Bu);
    gate_gumbel<<<Nn, 256, 0, stream>>>(acoustic, Wag, bag, acprob,  D, 8, 0.5f, 0xC2B2AE35u);
    make_masks<<<(Nn + 255) / 256, 256, 0, stream>>>(hiprob, capmask, acmask, Nn);

    // ---- 6. channel expert groups: y = sum_e probs_e*mask * (silu(xW1)*xW3)W2 ----
    for (int g = 0; g < 2; ++g) {
        const bf16_t* w1 = g ? aw1b : cw1b;
        const bf16_t* w2 = g ? aw2b : cw2b;
        const bf16_t* w3 = g ? aw3b : cw3b;
        const float*  pr = g ? acprob : capprob;
        const float*  mk = g ? acmask : capmask;
        for (int e = 0; e < E; ++e) {
            launch_gemm(stream, xb, w1 + (size_t)e * H * D, t1, nullptr,
                        Nn, H, D, D, D, H, 1, 1, 0,0,0,0,0,0, EPI_F32, 0, 0, 0, 0, 0);
            launch_gemm(stream, xb, w3 + (size_t)e * H * D, t3, nullptr,
                        Nn, H, D, D, D, H, 1, 1, 0,0,0,0,0,0, EPI_F32, 0, 0, 0, 0, 0);
            silu_mul_kernel<<<(unsigned)(((long)Nn * H + 255) / 256), 256, 0, stream>>>(
                t1, t3, hb, (long)Nn * H);
            float beta = (g == 0 && e == 0) ? 0.0f : 1.0f;
            launch_gemm(stream, hb, w2 + (size_t)e * D * H, y, nullptr,
                        Nn, D, H, H, H, D, 1, 1, 0,0,0,0,0,0,
                        EPI_ROWSCALE_ACC, beta, pr, E, e, mk);
        }
    }

    // ---- 7. feature experts (disjoint 96-channel slices; K reduced 768->96) ----
    launch_cvt(stream, y, yb, (long)Nn * D);
    for (int e = 0; e < E; ++e) {
        const bf16_t* Ae = yb + (size_t)e * HD;                       // lda = D
        launch_gemm(stream, Ae, fw1b + (size_t)e * H * D + (size_t)e * HD, t1, nullptr,
                    Nn, H, HD, D, D, H, 1, 1, 0,0,0,0,0,0, EPI_F32, 0, 0, 0, 0, 0);
        launch_gemm(stream, Ae, fw3b + (size_t)e * H * D + (size_t)e * HD, t3, nullptr,
                    Nn, H, HD, D, D, H, 1, 1, 0,0,0,0,0,0, EPI_F32, 0, 0, 0, 0, 0);
        silu_mul_kernel<<<(unsigned)(((long)Nn * H + 255) / 256), 256, 0, stream>>>(
            t1, t3, hb, (long)Nn * H);
        launch_gemm(stream, hb, fw2b + (size_t)e * D * H + (size_t)(e * HD) * H,
                    zout + (size_t)e * HD, nullptr,
                    Nn, HD, H, H, H, D, 1, 1, 0,0,0,0,0,0, EPI_F32, 0, 0, 0, 0, 0);
    }

    // ---- 8. load-balance loss ----
    loss_kernel<<<1, 256, 0, stream>>>(capprob, acprob, capmask, acmask, Nn,
                                       zout + (size_t)Nn * D);
}
